// MultiHeadSelfAttention_41120016891904
// MI455X (gfx1250) — compile-verified, matrix operand order checked by reading, not run on hardware
//
#include <hip/hip_runtime.h>
#include <hip/hip_bf16.h>

typedef __attribute__((ext_vector_type(16))) _Float16 v16h;
typedef __attribute__((ext_vector_type(8)))  float    v8f;
typedef __attribute__((ext_vector_type(4)))  int      v4i;

#define EMBED 1024
#define HALF  512
#define HEADS 16
#define DH    32
#define ATT_SCALE 0.125f   // 1/sqrt(EMBED/HEADS) = 1/8

#if __has_builtin(__builtin_amdgcn_global_load_async_to_lds_b128)
#define HAVE_ASYNC_LDS 1
#endif

// async-to-LDS builtin signature (probe-confirmed by diagnostics):
//   (v4i addrspace(1)* gsrc, v4i addrspace(3)* ldst, imm offset, imm cpol)
#define GPTR128(p) ((__attribute__((address_space(1))) v4i*)(p))
#define LPTR128(p) ((__attribute__((address_space(3))) v4i*)(p))

__device__ __forceinline__ void wait_async0() {
#if __has_builtin(__builtin_amdgcn_s_wait_asynccnt)
    __builtin_amdgcn_s_wait_asynccnt(0);
#else
    asm volatile("s_wait_asynccnt 0x0" ::: "memory");
#endif
}

// ---------------------------------------------------------------------------
// Fragment loaders per CDNA5 WMMA VGPR layouts (CDNA5 ISA §7.12.2)
// ---------------------------------------------------------------------------

// A-matrix 16x32 f16, row-major source with stride lda.
__device__ __forceinline__ v16h load_A_frag(const _Float16* __restrict__ A, int lda) {
    int l  = threadIdx.x & 31;
    int m  = l & 15;
    int kb = (l >> 4) * 8;
    const _Float16* row = A + m * lda;
    v16h a;
#pragma unroll
    for (int e = 0; e < 8; ++e) {
        a[e]     = row[kb + e];
        a[e + 8] = row[kb + 16 + e];
    }
    return a;
}

// B-matrix 32x16 where B[k][n] = W[n][k] (W stored (N,K) row-major, stride ldw).
__device__ __forceinline__ v16h load_B_from_WT(const _Float16* __restrict__ W, int ldw) {
    int l  = threadIdx.x & 31;
    int n  = l & 15;
    int kb = (l >> 4) * 8;
    const _Float16* row = W + n * ldw;
    v16h b;
#pragma unroll
    for (int e = 0; e < 8; ++e) {
        b[e]     = row[kb + e];
        b[e + 8] = row[kb + 16 + e];
    }
    return b;
}

// B-matrix 32x16 where B[k][n] = V[k*ldv + n] (row-major K x N source).
__device__ __forceinline__ v16h load_B_rowmajor(const _Float16* __restrict__ V, int ldv) {
    int l  = threadIdx.x & 31;
    int n  = l & 15;
    int kb = (l >> 4) * 8;
    v16h b;
#pragma unroll
    for (int e = 0; e < 8; ++e) {
        b[e]     = V[(kb + e) * ldv + n];
        b[e + 8] = V[(kb + 16 + e) * ldv + n];
    }
    return b;
}

// ---------------------------------------------------------------------------
// fp32 -> f16 conversion kernels
// ---------------------------------------------------------------------------
__global__ void cvt_left_half_f16(const float* __restrict__ x,
                                  _Float16* __restrict__ xh, int total) {
    int i = blockIdx.x * blockDim.x + threadIdx.x;
    if (i < total) {
        int s = i >> 9;
        int c = i & (HALF - 1);
        xh[i] = (_Float16)x[s * EMBED + c];
    }
}

__global__ void cvt_f32_to_f16(const float* __restrict__ src,
                               _Float16* __restrict__ dst, int n) {
    int i = blockIdx.x * blockDim.x + threadIdx.x;
    if (i < n) dst[i] = (_Float16)src[i];
}

// ---------------------------------------------------------------------------
// GEMM: C[M][N] = A[M][K] @ W[N][K]^T   (f16 in, f16 out), 64x32 tile/wave
// ---------------------------------------------------------------------------
__global__ void gemm_wT_f16(const _Float16* __restrict__ A,
                            const _Float16* __restrict__ W,
                            _Float16* __restrict__ C,
                            int M, int N, int K) {
    int w       = (blockIdx.x * blockDim.x + threadIdx.x) >> 5;
    int tilesN  = N / 32;
    int mt      = w / tilesN;
    int nt      = w % tilesN;
    if (mt * 64 >= M) return;
    int m0 = mt * 64, n0 = nt * 32;

    v8f acc[4][2];
#pragma unroll
    for (int i = 0; i < 4; ++i)
#pragma unroll
        for (int j = 0; j < 2; ++j) acc[i][j] = (v8f){};

    for (int k0 = 0; k0 < K; k0 += 32) {
        if (k0 + 32 < K) {
            __builtin_prefetch(A + m0 * K + k0 + 32, 0, 3);
            __builtin_prefetch(W + n0 * K + k0 + 32, 0, 3);
        }
        v16h af[4], bf[2];
#pragma unroll
        for (int i = 0; i < 4; ++i) af[i] = load_A_frag(A + (m0 + 16 * i) * K + k0, K);
#pragma unroll
        for (int j = 0; j < 2; ++j) bf[j] = load_B_from_WT(W + (n0 + 16 * j) * K + k0, K);
#pragma unroll
        for (int i = 0; i < 4; ++i)
#pragma unroll
            for (int j = 0; j < 2; ++j)
                acc[i][j] = __builtin_amdgcn_wmma_f32_16x16x32_f16(
                    false, af[i], false, bf[j], (short)0, acc[i][j], false, false);
    }

    int l  = threadIdx.x & 31;
    int n  = l & 15;
    int mb = (l >> 4) * 8;
#pragma unroll
    for (int i = 0; i < 4; ++i)
#pragma unroll
        for (int j = 0; j < 2; ++j)
#pragma unroll
            for (int r = 0; r < 8; ++r)
                C[(m0 + 16 * i + mb + r) * N + n0 + 16 * j + n] = (_Float16)acc[i][j][r];
}

// ---------------------------------------------------------------------------
// GEMM + bias, fp32 out
// ---------------------------------------------------------------------------
__global__ void gemm_wT_f32_bias(const _Float16* __restrict__ A,
                                 const _Float16* __restrict__ W,
                                 const float* __restrict__ bias,
                                 float* __restrict__ C,
                                 int M, int N, int K) {
    int w       = (blockIdx.x * blockDim.x + threadIdx.x) >> 5;
    int tilesN  = N / 32;
    int mt      = w / tilesN;
    int nt      = w % tilesN;
    if (mt * 64 >= M) return;
    int m0 = mt * 64, n0 = nt * 32;

    v8f acc[4][2];
#pragma unroll
    for (int i = 0; i < 4; ++i)
#pragma unroll
        for (int j = 0; j < 2; ++j) acc[i][j] = (v8f){};

    for (int k0 = 0; k0 < K; k0 += 32) {
        v16h af[4], bf[2];
#pragma unroll
        for (int i = 0; i < 4; ++i) af[i] = load_A_frag(A + (m0 + 16 * i) * K + k0, K);
#pragma unroll
        for (int j = 0; j < 2; ++j) bf[j] = load_B_from_WT(W + (n0 + 16 * j) * K + k0, K);
#pragma unroll
        for (int i = 0; i < 4; ++i)
#pragma unroll
            for (int j = 0; j < 2; ++j)
                acc[i][j] = __builtin_amdgcn_wmma_f32_16x16x32_f16(
                    false, af[i], false, bf[j], (short)0, acc[i][j], false, false);
    }

    int l  = threadIdx.x & 31;
    int n  = l & 15;
    int mb = (l >> 4) * 8;
#pragma unroll
    for (int i = 0; i < 4; ++i)
#pragma unroll
        for (int j = 0; j < 2; ++j) {
            float b = bias[n0 + 16 * j + n];
#pragma unroll
            for (int r = 0; r < 8; ++r)
                C[(m0 + 16 * i + mb + r) * N + n0 + 16 * j + n] = acc[i][j][r] + b;
        }
}

// ---------------------------------------------------------------------------
// Flash attention, block-cooperative:
//   block = 4 waves = 128 threads, one head, 128 query rows (32 per wave).
//   Per 32-key step: K/V tiles (32x32 f16 each) staged to LDS once per block
//   via global_load_async_to_lds_b128 (ASYNCcnt), then each wave runs
//   4 WMMAs for S = Q K^T, online softmax, 4 WMMAs for O += P V.
// ---------------------------------------------------------------------------
#define ATT_WAVES 4

__global__ void attention_fwd(const _Float16* __restrict__ Q,
                              const _Float16* __restrict__ K,
                              const _Float16* __restrict__ V,
                              _Float16* __restrict__ O, int S) {
    __shared__ _Float16 Ktile[32 * 32];             // [key][d]
    __shared__ _Float16 Vtile[32 * 32];             // [key][d]
    __shared__ _Float16 Pb[ATT_WAVES][32 * 32];     // per-wave P bounce [row][key]

    int wInBlk  = threadIdx.x >> 5;
    int qblocks = S / (ATT_WAVES * 32);
    int h       = blockIdx.x / qblocks;
    int qb      = blockIdx.x % qblocks;
    if (h >= HEADS) return;
    int q0   = qb * (ATT_WAVES * 32) + wInBlk * 32;  // this wave's 32 query rows
    int hoff = h * DH;

    int l  = threadIdx.x & 31;
    int n  = l & 15;
    int mb = (l >> 4) * 8;
    _Float16* myP = &Pb[wInBlk][0];

    v16h a_q[2];
    a_q[0] = load_A_frag(Q + q0 * HALF + hoff, HALF);
    a_q[1] = load_A_frag(Q + (q0 + 16) * HALF + hoff, HALF);

    v8f acc[2][2];
    float m_run[2][8], l_run[2][8];
#pragma unroll
    for (int i = 0; i < 2; ++i) {
#pragma unroll
        for (int t = 0; t < 2; ++t) acc[i][t] = (v8f){};
#pragma unroll
        for (int r = 0; r < 8; ++r) { m_run[i][r] = -3.0e38f; l_run[i][r] = 0.0f; }
    }

    // staging coords: 128 threads cover 32 rows x 4 x 16B chunks per tile
    int srow = threadIdx.x >> 2;
    int schk = (threadIdx.x & 3) * 8;   // f16 elements

    for (int j = 0; j < S; j += 32) {
        // ---- stage K/V 32x32 tiles into LDS (async, ASYNCcnt-tracked) ----
        {
            const _Float16* ksrc = K + (j + srow) * HALF + hoff + schk;
            const _Float16* vsrc = V + (j + srow) * HALF + hoff + schk;
            _Float16* kdst = &Ktile[srow * 32 + schk];
            _Float16* vdst = &Vtile[srow * 32 + schk];
#ifdef HAVE_ASYNC_LDS
            __builtin_amdgcn_global_load_async_to_lds_b128(GPTR128(ksrc), LPTR128(kdst), 0, 0);
            __builtin_amdgcn_global_load_async_to_lds_b128(GPTR128(vsrc), LPTR128(vdst), 0, 0);
            wait_async0();
#else
            *(uint4*)kdst = *(const uint4*)ksrc;
            *(uint4*)vdst = *(const uint4*)vsrc;
#endif
        }
        __syncthreads();

        // ---- S = Q * K^T : B[d][key] = Ktile[key][d] (row-contiguous d) ----
        v16h b_k0 = load_B_from_WT(&Ktile[0],       32);
        v16h b_k1 = load_B_from_WT(&Ktile[16 * 32], 32);
        v8f z = (v8f){};
        v8f s[2][2];
#pragma unroll
        for (int i = 0; i < 2; ++i) {
            s[i][0] = __builtin_amdgcn_wmma_f32_16x16x32_f16(false, a_q[i], false, b_k0,
                                                             (short)0, z, false, false);
            s[i][1] = __builtin_amdgcn_wmma_f32_16x16x32_f16(false, a_q[i], false, b_k1,
                                                             (short)0, z, false, false);
        }

        // ---- online softmax per 16-row tile; P -> per-wave LDS bounce ----
#pragma unroll
        for (int i = 0; i < 2; ++i) {
#pragma unroll
            for (int r = 0; r < 8; ++r) {
                float v0 = s[i][0][r] * ATT_SCALE;
                float v1 = s[i][1][r] * ATT_SCALE;
                float rm = fmaxf(v0, v1);
#pragma unroll
                for (int msk = 1; msk < 16; msk <<= 1)
                    rm = fmaxf(rm, __shfl_xor(rm, msk, 32));
                float m_new = fmaxf(m_run[i][r], rm);
                float corr  = __expf(m_run[i][r] - m_new);
                float p0    = __expf(v0 - m_new);
                float p1    = __expf(v1 - m_new);
                float rs    = p0 + p1;
#pragma unroll
                for (int msk = 1; msk < 16; msk <<= 1)
                    rs += __shfl_xor(rs, msk, 32);
                l_run[i][r] = l_run[i][r] * corr + rs;
                m_run[i][r] = m_new;
                acc[i][0][r] *= corr;
                acc[i][1][r] *= corr;
                myP[(i * 16 + mb + r) * 32 + n]      = (_Float16)p0;
                myP[(i * 16 + mb + r) * 32 + n + 16] = (_Float16)p1;
            }
        }

        // ---- reload P as A-fragments; O += P * V ----
        v16h b_v0 = load_B_rowmajor(&Vtile[0],  32);
        v16h b_v1 = load_B_rowmajor(&Vtile[16], 32);
#pragma unroll
        for (int i = 0; i < 2; ++i) {
            v16h a_p;
            {
                int am = l & 15;
                int kb = (l >> 4) * 8;
                const _Float16* prow = &myP[(i * 16 + am) * 32];
#pragma unroll
                for (int e = 0; e < 8; ++e) {
                    a_p[e]     = prow[kb + e];
                    a_p[e + 8] = prow[kb + 16 + e];
                }
            }
            acc[i][0] = __builtin_amdgcn_wmma_f32_16x16x32_f16(false, a_p, false, b_v0,
                                                               (short)0, acc[i][0], false, false);
            acc[i][1] = __builtin_amdgcn_wmma_f32_16x16x32_f16(false, a_p, false, b_v1,
                                                               (short)0, acc[i][1], false, false);
        }
        __syncthreads();   // protect K/V tiles before next stage
    }

    // ---- normalize and store O (f16, concat-head layout S x HALF) ----
#pragma unroll
    for (int i = 0; i < 2; ++i)
#pragma unroll
        for (int r = 0; r < 8; ++r) {
            float inv = 1.0f / l_run[i][r];
            int row = q0 + i * 16 + mb + r;
            O[row * HALF + hoff + n]      = (_Float16)(acc[i][0][r] * inv);
            O[row * HALF + hoff + 16 + n] = (_Float16)(acc[i][1][r] * inv);
        }
}

// ---------------------------------------------------------------------------
// Host-side launcher
// ---------------------------------------------------------------------------
extern "C" void kernel_launch(void* const* d_in, const int* in_sizes, int n_in,
                              void* d_out, int out_size, void* d_ws, size_t ws_size,
                              hipStream_t stream) {
    const float* x  = (const float*)d_in[0];   // S x EMBED
    const float* Wq = (const float*)d_in[1];   // HALF x HALF
    const float* Wk = (const float*)d_in[2];
    const float* Wv = (const float*)d_in[3];
    const float* Wo = (const float*)d_in[4];   // EMBED x HALF
    const float* bo = (const float*)d_in[5];   // EMBED
    float* out = (float*)d_out;                // S x EMBED

    const int S = in_sizes[0] / EMBED;         // 4096

    char* ws = (char*)d_ws;
    _Float16* Xh   = (_Float16*)ws;                  ws += (size_t)S * HALF * 2;
    _Float16* Wq16 = (_Float16*)ws;                  ws += (size_t)HALF * HALF * 2;
    _Float16* Wk16 = (_Float16*)ws;                  ws += (size_t)HALF * HALF * 2;
    _Float16* Wv16 = (_Float16*)ws;                  ws += (size_t)HALF * HALF * 2;
    _Float16* Wo16 = (_Float16*)ws;                  ws += (size_t)EMBED * HALF * 2;
    _Float16* Qm   = (_Float16*)ws;                  ws += (size_t)S * HALF * 2;
    _Float16* Km   = (_Float16*)ws;                  ws += (size_t)S * HALF * 2;
    _Float16* Vm   = (_Float16*)ws;                  ws += (size_t)S * HALF * 2;
    _Float16* Om   = (_Float16*)ws;

    // 1) conversions
    {
        int n = S * HALF;
        cvt_left_half_f16<<<(n + 255) / 256, 256, 0, stream>>>(x, Xh, n);
        int nw = HALF * HALF;
        cvt_f32_to_f16<<<(nw + 255) / 256, 256, 0, stream>>>(Wq, Wq16, nw);
        cvt_f32_to_f16<<<(nw + 255) / 256, 256, 0, stream>>>(Wk, Wk16, nw);
        cvt_f32_to_f16<<<(nw + 255) / 256, 256, 0, stream>>>(Wv, Wv16, nw);
        int no = EMBED * HALF;
        cvt_f32_to_f16<<<(no + 255) / 256, 256, 0, stream>>>(Wo, Wo16, no);
    }

    // 2) Q/K/V projections: M=S, N=HALF, K=HALF
    {
        int waves  = (S / 64) * (HALF / 32);
        int blocks = (waves * 32 + 255) / 256;
        gemm_wT_f16<<<blocks, 256, 0, stream>>>(Xh, Wq16, Qm, S, HALF, HALF);
        gemm_wT_f16<<<blocks, 256, 0, stream>>>(Xh, Wk16, Km, S, HALF, HALF);
        gemm_wT_f16<<<blocks, 256, 0, stream>>>(Xh, Wv16, Vm, S, HALF, HALF);
    }

    // 3) attention: block = 4 waves, one head, 128 query rows
    {
        int blocks = HEADS * (S / (ATT_WAVES * 32));
        attention_fwd<<<blocks, ATT_WAVES * 32, 0, stream>>>(Qm, Km, Vm, Om, S);
    }

    // 4) output projection: M=S, N=EMBED, K=HALF, + bias, fp32 out
    {
        int waves  = (S / 64) * (EMBED / 32);
        int blocks = (waves * 32 + 255) / 256;
        gemm_wT_f32_bias<<<blocks, 256, 0, stream>>>(Om, Wo16, bo, out, S, EMBED, HALF);
    }
}
